// RiskAwareGATPolicy_1357209665763
// MI455X (gfx1250) — matrix operand order repaired and checked by reading, not compiled, not run on hardware
//
#include <hip/hip_runtime.h>
#include <hip/hip_bf16.h>
#include <math.h>

// ---------------------------------------------------------------------------
// RiskAwareGATPolicy for MI455X (gfx1250, wave32, WMMA)
//   - GATv2 layers: VALU + f32 atomics (bandwidth/atomic bound, no matrix shape)
//   - Layer-1 node transform  [N,64]@[64,64]   : v_wmma_f32_16x16x32_f16
//   - Edge MLP                [E,194]@[194,32] : v_wmma_f32_16x16x32_f16
//     (K padded 194->224 = 7 tiles of 32), fused relu + [32,1] epilogue via
//     C-fragment layout + shfl_xor reduction.
//   Epilogues use tile-level full/partial split so the common path has no
//   per-element exec-mask juggling.
// ---------------------------------------------------------------------------

typedef _Float16 f16;
typedef __attribute__((ext_vector_type(8)))  _Float16 v8h;
typedef __attribute__((ext_vector_type(16))) _Float16 v16h;
typedef __attribute__((ext_vector_type(8)))  float    v8f;

#define FDIM 64          // H*C = 2*32
#define NEG_SLOPE 0.2f

static __device__ __forceinline__ v16h cat16(v8h lo, v8h hi) {
  return __builtin_shufflevector(lo, hi, 0,1,2,3,4,5,6,7,8,9,10,11,12,13,14,15);
}
static __device__ __forceinline__ v8f wmma_f16(v16h a, v16h b, v8f c) {
  // D = A(16x32 f16) x B(32x16 f16) + C(16x16 f32)
  return __builtin_amdgcn_wmma_f32_16x16x32_f16(false, a, false, b, (short)0, c,
                                                false, false);
}

// float atomic-max via int/uint monotonic bit trick (init must be -inf)
static __device__ __forceinline__ void atomicMaxF(float* addr, float v) {
  if (v >= 0.f) atomicMax((int*)addr, __float_as_int(v));
  else          atomicMin((unsigned int*)addr, __float_as_uint(v));
}

// ---------------------------------------------------------------------------
__global__ void fill_f32(float* p, float v, int n) {
  int i = blockIdx.x * blockDim.x + threadIdx.x;
  if (i < n) p[i] = v;
}

// column sums of edge_attr [E,2] -> sums[2]  (self-loop attr = mean)
__global__ void ea_sum_kernel(const float* __restrict__ ea, float* sums, int E) {
  __shared__ float s0[256], s1[256];
  int tid = threadIdx.x;
  float a0 = 0.f, a1 = 0.f;
  for (int e = blockIdx.x * blockDim.x + tid; e < E; e += gridDim.x * blockDim.x) {
    a0 += ea[2*e]; a1 += ea[2*e + 1];
  }
  s0[tid] = a0; s1[tid] = a1; __syncthreads();
  for (int s = 128; s > 0; s >>= 1) {
    if (tid < s) { s0[tid] += s0[tid+s]; s1[tid] += s1[tid+s]; }
    __syncthreads();
  }
  if (tid == 0) { atomicAdd(&sums[0], s0[0]); atomicAdd(&sums[1], s1[0]); }
}

// layer-0 node transform: z = x[N,2] @ W[2,64] + b   (too skinny for WMMA)
__global__ void node_lin0(const float* __restrict__ x, const float* __restrict__ W,
                          const float* __restrict__ b, float* __restrict__ z, int N) {
  int i = blockIdx.x * blockDim.x + threadIdx.x;
  if (i >= N * FDIM) return;
  int n = i >> 6, c = i & 63;
  z[i] = fmaf(x[2*n], W[c], fmaf(x[2*n + 1], W[64 + c], b[c]));
}

// per-edge attention logits + segment max:
//   s = leaky_relu(z[dst]+z[src]+ea@We), alpha[h] = dot(s_h, att_h)
__global__ void edge_alpha(const int* __restrict__ ei, const float* __restrict__ ea,
                           const float* __restrict__ ea_sum, float invE,
                           const float* __restrict__ z, const float* __restrict__ We,
                           const float* __restrict__ att,
                           float* __restrict__ alpha, float* __restrict__ m,
                           int E, int Etot) {
  int e = blockIdx.x * blockDim.x + threadIdx.x;
  if (e >= Etot) return;
  int src, dst; float a0, a1;
  if (e < E) { src = ei[e]; dst = ei[E + e]; a0 = ea[2*e]; a1 = ea[2*e + 1]; }
  else       { src = dst = e - E; a0 = ea_sum[0] * invE; a1 = ea_sum[1] * invE; }
  const float4* zs = (const float4*)(z + (long)src * FDIM);
  const float4* zd = (const float4*)(z + (long)dst * FDIM);
  float acc0 = 0.f, acc1 = 0.f;
#pragma unroll
  for (int q = 0; q < 16; ++q) {
    float4 vs = zs[q], vd = zd[q];
#pragma unroll
    for (int j = 0; j < 4; ++j) {
      int c = q * 4 + j;
      float s = ((const float*)&vs)[j] + ((const float*)&vd)[j]
              + fmaf(a0, We[c], a1 * We[64 + c]);
      s = (s >= 0.f) ? s : NEG_SLOPE * s;
      float t = s * att[c];
      if (c < 32) acc0 += t; else acc1 += t;
    }
  }
  alpha[2*e] = acc0; alpha[2*e + 1] = acc1;
  atomicMaxF(&m[2*dst], acc0);
  atomicMaxF(&m[2*dst + 1], acc1);
}

// ex = exp(alpha - m[dst]); denom[dst] += ex  (every node has a self loop -> m finite)
__global__ void edge_exp(const int* __restrict__ ei, float* __restrict__ alpha,
                         const float* __restrict__ m, float* __restrict__ denom,
                         int E, int Etot) {
  int e = blockIdx.x * blockDim.x + threadIdx.x;
  if (e >= Etot) return;
  int dst = (e < E) ? ei[E + e] : e - E;
  float e0 = expf(alpha[2*e]     - m[2*dst]);
  float e1 = expf(alpha[2*e + 1] - m[2*dst + 1]);
  alpha[2*e] = e0; alpha[2*e + 1] = e1;
  atomicAdd(&denom[2*dst], e0);
  atomicAdd(&denom[2*dst + 1], e1);
}

// out[dst] += z[src] * (ex / denom[dst])
__global__ void edge_aggregate(const int* __restrict__ ei, const float* __restrict__ alpha,
                               const float* __restrict__ denom, const float* __restrict__ z,
                               float* __restrict__ out, int E, int Etot) {
  int e = blockIdx.x * blockDim.x + threadIdx.x;
  if (e >= Etot) return;
  int src, dst;
  if (e < E) { src = ei[e]; dst = ei[E + e]; } else { src = dst = e - E; }
  float a0 = alpha[2*e]     / (denom[2*dst]     + 1e-16f);
  float a1 = alpha[2*e + 1] / (denom[2*dst + 1] + 1e-16f);
  const float4* zs = (const float4*)(z + (long)src * FDIM);
  float* o = out + (long)dst * FDIM;
#pragma unroll
  for (int q = 0; q < 16; ++q) {
    float4 v = zs[q];
    float a = (q < 8) ? a0 : a1;
#pragma unroll
    for (int j = 0; j < 4; ++j)
      atomicAdd(&o[q * 4 + j], ((const float*)&v)[j] * a);
  }
}

// h = elu(accum + bias), stored as f16 (WMMA feed for next GEMM)
__global__ void node_finalize(const float* __restrict__ accum, const float* __restrict__ bias,
                              f16* __restrict__ h16, int N) {
  int i = blockIdx.x * blockDim.x + threadIdx.x;
  if (i >= N * FDIM) return;
  float v = accum[i] + bias[i & 63];
  v = (v > 0.f) ? v : expf(v) - 1.f;
  h16[i] = (f16)v;
}

// Pre-swizzle W[K,Ncols] (f32, row-major) into wave32 B-fragment layout:
//   frag[((kt*NT+nt)*32 + lane)*16 + i]  with  k = kt*32 + (lane<16?0:8) + (i<8 ? i : i+8),
//   n = nt*16 + (lane&15); zero-padded for k >= K (K=194 -> 7 tiles).
__global__ void prep_bfrag(const float* __restrict__ W, f16* __restrict__ frag,
                           int K, int Ncols, int KT, int NT) {
  int i = blockIdx.x * blockDim.x + threadIdx.x;
  if (i >= KT * NT * 32 * 16) return;
  int idx  = i & 15;
  int lane = (i >> 4) & 31;
  int fi   = i >> 9;
  int nt   = fi % NT, kt = fi / NT;
  int n = nt * 16 + (lane & 15);
  int k = kt * 32 + ((lane < 16) ? 0 : 8) + (idx < 8 ? idx : idx + 8);
  frag[i] = (k < K) ? (f16)W[k * Ncols + n] : (f16)0.f;
}

// z = h16[N,64] @ W1[64,64] + b1  via WMMA; 4 waves/block, 16 rows/wave.
// A fragments loaded straight from global: lane (l&15) owns row, chunks of
// 8 contiguous halves at K = kt*32 + (l<16?0:8) and +16 (ISA A-layout).
__global__ void __launch_bounds__(128)
node_lin_wmma(const f16* __restrict__ h16, const f16* __restrict__ Bfrag,
              const float* __restrict__ b, float* __restrict__ z, int N) {
  const int lane = threadIdx.x & 31;
  const int wave = threadIdx.x >> 5;
  const int row0 = (blockIdx.x * 4 + wave) * 16;
  const int hs   = lane >> 4;                 // 0: lanes 0-15, 1: lanes 16-31
  const bool full = (row0 + 16 <= N);
  const int arow = full ? (row0 + (lane & 15)) : min(row0 + (lane & 15), N - 1);
  const f16* ar = h16 + (long)arow * FDIM;

  v8f acc[4] = {};
#pragma unroll
  for (int kt = 0; kt < 2; ++kt) {
    const int ka = kt * 32 + hs * 8;
    v16h A = cat16(*(const v8h*)(ar + ka), *(const v8h*)(ar + ka + 16));
#pragma unroll
    for (int nt = 0; nt < 4; ++nt) {
      v16h B = *(const v16h*)(Bfrag + ((kt * 4 + nt) * 32 + lane) * 16);
      acc[nt] = wmma_f16(A, B, acc[nt]);
    }
  }
  // C layout: VGPR v -> row v (lanes 0-15) / 8+v (lanes 16-31), col = lane&15
  const int rbase = row0 + hs * 8;
  if (full) {                     // common path: no per-element exec masking
#pragma unroll
    for (int nt = 0; nt < 4; ++nt) {
      int col = nt * 16 + (lane & 15);
      float bb = b[col];
      float* zp = z + (long)rbase * FDIM + col;
#pragma unroll
      for (int v = 0; v < 8; ++v)
        zp[(long)v * FDIM] = acc[nt][v] + bb;
    }
  } else {                        // last partial tile only
#pragma unroll
    for (int nt = 0; nt < 4; ++nt) {
      int col = nt * 16 + (lane & 15);
      float bb = b[col];
#pragma unroll
      for (int v = 0; v < 8; ++v)
        if (rbase + v < N) z[(long)(rbase + v) * FDIM + col] = acc[nt][v] + bb;
    }
  }
}

// Edge MLP: logits = relu([src,dst,goal,ea] @ Wm1 + bm1) @ Wm2 + bm2
// A row (194 cols padded to 224) gathered chunk-wise from h16; B is
// pre-swizzled Wm1. Fused epilogue: relu + Wm2 dot via shfl_xor reduce.
static __device__ __forceinline__ v8h gat_chunk(const f16* s, const f16* d,
                                                const f16* g, float ea0, float ea1,
                                                int c) {
  if (c < 64)  return *(const v8h*)(s + c);
  if (c < 128) return *(const v8h*)(d + c - 64);
  if (c < 192) return *(const v8h*)(g + c - 128);
  v8h r = {};
  if (c == 192) { r[0] = (f16)ea0; r[1] = (f16)ea1; }
  return r;
}

__global__ void __launch_bounds__(256)
edge_mlp_wmma(const int* __restrict__ ei, const float* __restrict__ ea,
              const f16* __restrict__ h16, const int* __restrict__ dest_idx,
              const f16* __restrict__ Bfrag, const float* __restrict__ bm1,
              const float* __restrict__ wm2, const float* __restrict__ bm2,
              float* __restrict__ logits, int E) {
  const int lane = threadIdx.x & 31;
  const int wave = threadIdx.x >> 5;
  const int e0   = (blockIdx.x * 8 + wave) * 16;
  const int hs   = lane >> 4;
  const bool full = (e0 + 16 <= E);
  const int ec   = full ? (e0 + (lane & 15)) : min(e0 + (lane & 15), E - 1);

  const int srcn = ei[ec], dstn = ei[E + ec], goal = *dest_idx;
  const float ea0 = ea[2*ec], ea1 = ea[2*ec + 1];
  const f16* as = h16 + (long)srcn * FDIM;
  const f16* ad = h16 + (long)dstn * FDIM;
  const f16* ag = h16 + (long)goal * FDIM;

  v8f acc0 = {}, acc1 = {};
#pragma unroll
  for (int kt = 0; kt < 7; ++kt) {
    const int c0 = kt * 32 + hs * 8;
    v16h A = cat16(gat_chunk(as, ad, ag, ea0, ea1, c0),
                   gat_chunk(as, ad, ag, ea0, ea1, c0 + 16));
    v16h B0 = *(const v16h*)(Bfrag + ((kt * 2 + 0) * 32 + lane) * 16);
    v16h B1 = *(const v16h*)(Bfrag + ((kt * 2 + 1) * 32 + lane) * 16);
    acc0 = wmma_f16(A, B0, acc0);
    acc1 = wmma_f16(A, B1, acc1);
  }

  const int n   = lane & 15;
  const float b1a = bm1[n], b1b = bm1[16 + n];
  const float w2a = wm2[n], w2b = wm2[16 + n];
  const float b2  = bm2[0];
  float res[8];
#pragma unroll
  for (int v = 0; v < 8; ++v) {
    float p = fmaxf(acc0[v] + b1a, 0.f) * w2a + fmaxf(acc1[v] + b1b, 0.f) * w2b;
    p += __shfl_xor(p, 1, 32);
    p += __shfl_xor(p, 2, 32);
    p += __shfl_xor(p, 4, 32);
    p += __shfl_xor(p, 8, 32);
    res[v] = p + b2;
  }
  // lane 0 holds rows e0..e0+7, lane 16 holds rows e0+8..e0+15
  if (n == 0) {
    const int rb = e0 + hs * 8;
    if (full) {
#pragma unroll
      for (int v = 0; v < 8; ++v) logits[rb + v] = res[v];
    } else {
#pragma unroll
      for (int v = 0; v < 8; ++v)
        if (rb + v < E) logits[rb + v] = res[v];
    }
  }
}

// ---------------------------------------------------------------------------
static inline size_t align256(size_t x) { return (x + 255) & ~(size_t)255; }

extern "C" void kernel_launch(void* const* d_in, const int* in_sizes, int n_in,
                              void* d_out, int out_size, void* d_ws, size_t ws_size,
                              hipStream_t stream) {
  const float* x    = (const float*)d_in[0];
  const int*   ei   = (const int*)  d_in[1];
  const float* ea   = (const float*)d_in[2];
  const int*   dest = (const int*)  d_in[3];
  const float* W0   = (const float*)d_in[4];
  const float* b0   = (const float*)d_in[5];
  const float* We0  = (const float*)d_in[6];
  const float* att0 = (const float*)d_in[7];
  const float* bia0 = (const float*)d_in[8];
  const float* W1   = (const float*)d_in[9];
  const float* b1   = (const float*)d_in[10];
  const float* We1  = (const float*)d_in[11];
  const float* att1 = (const float*)d_in[12];
  const float* bia1 = (const float*)d_in[13];
  const float* Wm1  = (const float*)d_in[14];
  const float* bm1  = (const float*)d_in[15];
  const float* Wm2  = (const float*)d_in[16];
  const float* bm2  = (const float*)d_in[17];
  float* logits = (float*)d_out;

  const int N = in_sizes[0] / 2;
  const int E = in_sizes[1] / 2;
  const int Etot = E + N;

  // workspace carve-up
  char* p = (char*)d_ws;
  float* z      = (float*)p; p += align256((size_t)N * FDIM * 4);
  float* accum  = (float*)p; p += align256((size_t)N * FDIM * 4);
  f16*   h16    = (f16*)  p; p += align256((size_t)N * FDIM * 2);
  float* alpha  = (float*)p; p += align256((size_t)Etot * 2 * 4);
  float* m      = (float*)p; p += align256((size_t)N * 2 * 4);
  float* denom  = (float*)p; p += align256((size_t)N * 2 * 4);
  float* easum  = (float*)p; p += align256(2 * 4);
  f16*   fragW1 = (f16*)  p; p += align256((size_t)2 * 4 * 32 * 16 * 2);
  f16*   fragWm = (f16*)  p; p += align256((size_t)7 * 2 * 32 * 16 * 2);
  (void)ws_size; (void)n_in; (void)out_size;

  const int T = 256;
  const int gNF   = (N * FDIM + T - 1) / T;
  const int gE    = (Etot + T - 1) / T;
  const int gN2   = (N * 2 + T - 1) / T;

  // one-time (per call) prep: edge-attr mean + B-fragment swizzles
  fill_f32<<<1, 32, 0, stream>>>(easum, 0.f, 2);
  ea_sum_kernel<<<640, 256, 0, stream>>>(ea, easum, E);
  prep_bfrag<<<(2*4*32*16 + T - 1)/T, T, 0, stream>>>(W1,  fragW1, 64,  64, 2, 4);
  prep_bfrag<<<(7*2*32*16 + T - 1)/T, T, 0, stream>>>(Wm1, fragWm, 194, 32, 7, 2);

  // ---- layer 0 ----
  node_lin0<<<gNF, T, 0, stream>>>(x, W0, b0, z, N);
  fill_f32<<<gN2, T, 0, stream>>>(m, -INFINITY, N * 2);
  fill_f32<<<gN2, T, 0, stream>>>(denom, 0.f, N * 2);
  fill_f32<<<gNF, T, 0, stream>>>(accum, 0.f, N * FDIM);
  edge_alpha<<<gE, T, 0, stream>>>(ei, ea, easum, 1.f / (float)E, z, We0, att0,
                                   alpha, m, E, Etot);
  edge_exp<<<gE, T, 0, stream>>>(ei, alpha, m, denom, E, Etot);
  edge_aggregate<<<gE, T, 0, stream>>>(ei, alpha, denom, z, accum, E, Etot);
  node_finalize<<<gNF, T, 0, stream>>>(accum, bia0, h16, N);

  // ---- layer 1 (node transform on WMMA) ----
  node_lin_wmma<<<(N + 63) / 64, 128, 0, stream>>>(h16, fragW1, b1, z, N);
  fill_f32<<<gN2, T, 0, stream>>>(m, -INFINITY, N * 2);
  fill_f32<<<gN2, T, 0, stream>>>(denom, 0.f, N * 2);
  fill_f32<<<gNF, T, 0, stream>>>(accum, 0.f, N * FDIM);
  edge_alpha<<<gE, T, 0, stream>>>(ei, ea, easum, 1.f / (float)E, z, We1, att1,
                                   alpha, m, E, Etot);
  edge_exp<<<gE, T, 0, stream>>>(ei, alpha, m, denom, E, Etot);
  edge_aggregate<<<gE, T, 0, stream>>>(ei, alpha, denom, z, accum, E, Etot);
  node_finalize<<<gNF, T, 0, stream>>>(accum, bia1, h16, N);

  // ---- edge MLP (WMMA, fused epilogue) ----
  edge_mlp_wmma<<<(E + 127) / 128, 256, 0, stream>>>(ei, ea, h16, dest, fragWm,
                                                     bm1, Wm2, bm2, logits, E);
}
